// gNet_42889543417916
// MI455X (gfx1250) — matrix operand order repaired and checked by reading, not compiled, not run on hardware
//
#include <hip/hip_runtime.h>
#include <hip/hip_bf16.h>
#include <stdint.h>

typedef __bf16 v16bf __attribute__((ext_vector_type(16)));
typedef __bf16 v8bf  __attribute__((ext_vector_type(8)));
typedef float  v8f   __attribute__((ext_vector_type(8)));

#define LN_EPS 1e-5f

// ---------------------------------------------------------------------------
// Fused double GEMM:  out[O x C] = W2[O x H] * relu( W1[H x I] * X[I x C] )
// The H x 64 intermediate lives entirely in LDS as bf16 in [col][k] layout so
// phase-2 B-fragments are two contiguous ds_load_b128 per lane.
//
// GATHER : input rows [0,128) come from Xg[k*Ngather + iInd[col]], rows
//          [128,I) from X[(k-128)*C + col]   (edge-conv input concat).
// SCATTER: epilogue does edge_ave scatter (atomicAdd 0.5*y to xnc at iInd and
//          jInd) and the in-place residual xe = relu(xe + y). Nothing stored
//          to Y. Race-free: each block touches only its own 64 columns of xe.
// STATS  : epilogue stores Y and block-reduces sum/sumsq for tensor_ln.
// Block = 256 threads = 8 waves; 64-column tile; wave w -> col sub-tile w%4,
// row half w>>2 of each 32-row pass.
// ---------------------------------------------------------------------------
template<int I, int H, int O, bool GATHER, bool SCATTER, bool STATS>
__global__ __launch_bounds__(256)
void fused2_wmma(const float* __restrict__ W1,
                 const float* __restrict__ W2,
                 const float* __restrict__ X,
                 const float* __restrict__ Xg,
                 const int*   __restrict__ iInd,
                 const int*   __restrict__ jInd,
                 int C, int Ngather,
                 float* __restrict__ Y,
                 float* __restrict__ xe,
                 float* __restrict__ xnc,
                 float* __restrict__ stats)
{
    __shared__ __bf16 tileX[64 * 32];  // staged input tile, [c][k]
    __shared__ __bf16 midb[64 * H];    // relu(W1*X) tile, [c][m]

    const int tid   = threadIdx.x;
    const int lane  = tid & 31;
    const int wave  = tid >> 5;
    const int col0  = blockIdx.x * 64;
    const int wcol  = (wave & 3) * 16;
    const int rhalf = wave >> 2;
    constexpr int R1 = H / 32;
    constexpr int R2 = O / 32;

    // cooperative loader mapping: thread -> (column, 8 consecutive k rows)
    const int  lc    = tid & 63;
    const int  lk    = (tid >> 6) * 8;
    const int  gcol  = col0 + lc;
    const bool colok = gcol < C;
    int gidx = 0;
    if (GATHER && colok) gidx = iInd[gcol];

    // ---------------- phase 1: acc1 = W1 * X ----------------
    v8f acc1[R1];
#pragma unroll
    for (int r = 0; r < R1; ++r)
#pragma unroll
        for (int v = 0; v < 8; ++v) acc1[r][v] = 0.f;

    for (int k0 = 0; k0 < I; k0 += 32) {
        __bf16 vals[8];
#pragma unroll
        for (int u = 0; u < 8; ++u) {
            const int k = k0 + lk + u;
            float f = 0.f;
            if (colok) {
                if (GATHER) {
                    f = (k < 128) ? Xg[(size_t)k * Ngather + gidx]
                                  : X[(size_t)(k - 128) * C + gcol];
                } else {
                    f = X[(size_t)k * C + gcol];
                }
            }
            vals[u] = (__bf16)f;
        }
        __syncthreads();
#pragma unroll
        for (int u = 0; u < 8; ++u) tileX[lc * 32 + lk + u] = vals[u];
        __syncthreads();

        v16bf bf;
        const int bc = wcol + (lane & 15);
        const int kb = (lane < 16) ? 0 : 16;
#pragma unroll
        for (int u = 0; u < 16; ++u) bf[u] = tileX[bc * 32 + kb + u];

        const int ka = (lane < 16) ? 0 : 16;
#pragma unroll
        for (int r = 0; r < R1; ++r) {
            const int m = r * 32 + rhalf * 16 + (lane & 15);
            const float* wr = W1 + (size_t)m * I + k0;
            v16bf af;
#pragma unroll
            for (int u = 0; u < 8; ++u) af[u]     = (__bf16)wr[ka + u];
#pragma unroll
            for (int u = 0; u < 8; ++u) af[8 + u] = (__bf16)wr[ka + 8 + u];
            acc1[r] = __builtin_amdgcn_wmma_f32_16x16x32_bf16(
                          false, af, false, bf, (short)0, acc1[r], false, false);
        }
    }

    // relu + bf16 pack into midb[c][m]; 8 consecutive rows -> one 16B ds store
    {
        const int mcol = wcol + (lane & 15);
#pragma unroll
        for (int r = 0; r < R1; ++r) {
            const int mbase = r * 32 + rhalf * 16 + ((lane < 16) ? 0 : 8);
            v8bf pk;
#pragma unroll
            for (int v = 0; v < 8; ++v) pk[v] = (__bf16)fmaxf(acc1[r][v], 0.f);
            *reinterpret_cast<v8bf*>(&midb[mcol * H + mbase]) = pk;
        }
    }
    __syncthreads();

    // ---------------- phase 2: acc2 = W2 * mid ----------------
    v8f acc2[R2];
#pragma unroll
    for (int r = 0; r < R2; ++r)
#pragma unroll
        for (int v = 0; v < 8; ++v) acc2[r][v] = 0.f;

    for (int k0 = 0; k0 < H; k0 += 32) {
        v16bf bf;
        const int bc = wcol + (lane & 15);
        const int kb = (lane < 16) ? 0 : 16;
#pragma unroll
        for (int u = 0; u < 16; ++u) bf[u] = midb[bc * H + k0 + kb + u];

        const int ka = (lane < 16) ? 0 : 16;
#pragma unroll
        for (int r = 0; r < R2; ++r) {
            const int m = r * 32 + rhalf * 16 + (lane & 15);
            const float* wr = W2 + (size_t)m * H + k0;
            v16bf af;
#pragma unroll
            for (int u = 0; u < 8; ++u) af[u]     = (__bf16)wr[ka + u];
#pragma unroll
            for (int u = 0; u < 8; ++u) af[8 + u] = (__bf16)wr[ka + 8 + u];
            acc2[r] = __builtin_amdgcn_wmma_f32_16x16x32_bf16(
                          false, af, false, bf, (short)0, acc2[r], false, false);
        }
    }

    // ---------------- epilogue ----------------
    const int  ocol = col0 + wcol + (lane & 15);
    const bool ook  = ocol < C;

    if constexpr (SCATTER) {
        int di = 0, dj = 0;
        if (ook) { di = iInd[ocol]; dj = jInd[ocol]; }
#pragma unroll
        for (int r = 0; r < R2; ++r) {
            const int mbase = r * 32 + rhalf * 16 + ((lane < 16) ? 0 : 8);
#pragma unroll
            for (int v = 0; v < 8; ++v) {
                if (ook) {
                    const float y = acc2[r][v];
                    const int   m = mbase + v;
                    const float hv = 0.5f * y;
                    atomicAdd(&xnc[(size_t)m * Ngather + di], hv);
                    atomicAdd(&xnc[(size_t)m * Ngather + dj], hv);
                    const size_t off = (size_t)m * C + ocol;
                    xe[off] = fmaxf(xe[off] + y, 0.f);
                }
            }
        }
    } else {
        float lsum = 0.f, lsq = 0.f;
#pragma unroll
        for (int r = 0; r < R2; ++r) {
            const int mbase = r * 32 + rhalf * 16 + ((lane < 16) ? 0 : 8);
#pragma unroll
            for (int v = 0; v < 8; ++v) {
                const float y = acc2[r][v];          // OOB cols are exactly 0
                if (STATS) { lsum += y; lsq += y * y; }
                if (ook) Y[(size_t)(mbase + v) * C + ocol] = y;
            }
        }
        if constexpr (STATS) {
            __shared__ float s1[256], s2[256];
            s1[tid] = lsum; s2[tid] = lsq;
            __syncthreads();
            for (int s = 128; s > 0; s >>= 1) {
                if (tid < s) { s1[tid] += s1[tid + s]; s2[tid] += s2[tid + s]; }
                __syncthreads();
            }
            if (tid == 0) { atomicAdd(&stats[0], s1[0]); atomicAdd(&stats[1], s2[0]); }
        }
    }
}

// ---------------------------------------------------------------------------
// Single WMMA GEMM (final projection): Y[O x C] = W[O x I] * X[I x C]
// ---------------------------------------------------------------------------
template<int I, int O>
__global__ __launch_bounds__(256)
void gemm_wmma(const float* __restrict__ W, const float* __restrict__ X,
               int C, float* __restrict__ Y)
{
    __shared__ __bf16 tileX[64 * 32];
    const int tid   = threadIdx.x;
    const int lane  = tid & 31;
    const int wave  = tid >> 5;
    const int col0  = blockIdx.x * 64;
    const int wcol  = (wave & 3) * 16;
    const int rhalf = wave >> 2;
    constexpr int R = O / 32;

    v8f acc[R];
#pragma unroll
    for (int r = 0; r < R; ++r)
#pragma unroll
        for (int v = 0; v < 8; ++v) acc[r][v] = 0.f;

    const int  lc    = tid & 63;
    const int  lk    = (tid >> 6) * 8;
    const int  gcol  = col0 + lc;
    const bool colok = gcol < C;

    for (int k0 = 0; k0 < I; k0 += 32) {
        __bf16 vals[8];
#pragma unroll
        for (int u = 0; u < 8; ++u) {
            const int k = k0 + lk + u;
            vals[u] = (__bf16)(colok ? X[(size_t)k * C + gcol] : 0.f);
        }
        __syncthreads();
#pragma unroll
        for (int u = 0; u < 8; ++u) tileX[lc * 32 + lk + u] = vals[u];
        __syncthreads();

        v16bf bf;
        const int bc = wcol + (lane & 15);
        const int kb = (lane < 16) ? 0 : 16;
#pragma unroll
        for (int u = 0; u < 16; ++u) bf[u] = tileX[bc * 32 + kb + u];

        const int ka = (lane < 16) ? 0 : 16;
#pragma unroll
        for (int r = 0; r < R; ++r) {
            const int m = r * 32 + rhalf * 16 + (lane & 15);
            const float* wr = W + (size_t)m * I + k0;
            v16bf af;
#pragma unroll
            for (int u = 0; u < 8; ++u) af[u]     = (__bf16)wr[ka + u];
#pragma unroll
            for (int u = 0; u < 8; ++u) af[8 + u] = (__bf16)wr[ka + 8 + u];
            acc[r] = __builtin_amdgcn_wmma_f32_16x16x32_bf16(
                         false, af, false, bf, (short)0, acc[r], false, false);
        }
    }

    const int  ocol = col0 + wcol + (lane & 15);
    if (ocol < C) {
#pragma unroll
        for (int r = 0; r < R; ++r) {
            const int mbase = r * 32 + rhalf * 16 + ((lane < 16) ? 0 : 8);
#pragma unroll
            for (int v = 0; v < 8; ++v)
                Y[(size_t)(mbase + v) * C + ocol] = acc[r][v];
        }
    }
}

// ---------------------------------------------------------------------------
// helpers
// ---------------------------------------------------------------------------
__global__ void fillz(float* p, size_t n) {
    size_t i  = (size_t)blockIdx.x * blockDim.x + threadIdx.x;
    size_t st = (size_t)gridDim.x * blockDim.x;
    for (; i < n; i += st) p[i] = 0.f;
}

__global__ void finalize_stats(float* stats, float invcnt) {
    float m = stats[0] * invcnt;
    float v = stats[1] * invcnt - m * m;
    stats[2] = m;
    stats[3] = rsqrtf(v + LN_EPS);
}

__global__ void ln_apply(float* x, size_t n, const float* __restrict__ stats) {
    const float m = stats[2], r = stats[3];
    size_t i  = (size_t)blockIdx.x * blockDim.x + threadIdx.x;
    size_t st = (size_t)gridDim.x * blockDim.x;
    for (; i < n; i += st) x[i] = (x[i] - m) * r;
}

// xn = relu(xn + tensor_ln(xnc))
__global__ void node_update(float* xn, const float* __restrict__ xnc, size_t n,
                            const float* __restrict__ stats) {
    const float m = stats[2], r = stats[3];
    size_t i  = (size_t)blockIdx.x * blockDim.x + threadIdx.x;
    size_t st = (size_t)gridDim.x * blockDim.x;
    for (; i < n; i += st) xn[i] = fmaxf(xn[i] + (xnc[i] - m) * r, 0.f);
}

__global__ void reduce_stats(const float* __restrict__ x, size_t n, float* stats) {
    __shared__ float s1[256], s2[256];
    size_t i  = (size_t)blockIdx.x * blockDim.x + threadIdx.x;
    size_t st = (size_t)gridDim.x * blockDim.x;
    float a = 0.f, b = 0.f;
    for (; i < n; i += st) { float v = x[i]; a += v; b += v * v; }
    s1[threadIdx.x] = a; s2[threadIdx.x] = b;
    __syncthreads();
    for (int s = 128; s > 0; s >>= 1) {
        if ((int)threadIdx.x < s) {
            s1[threadIdx.x] += s1[threadIdx.x + s];
            s2[threadIdx.x] += s2[threadIdx.x + s];
        }
        __syncthreads();
    }
    if (threadIdx.x == 0) { atomicAdd(&stats[0], s1[0]); atomicAdd(&stats[1], s2[0]); }
}

// Fold KE1 (256x384 = [A1|A2|A3]) into Wf (256x256 = [A1+A2 | A3]) since row==col.
__global__ void fold_ke1(const float* __restrict__ KE1i, float* Wf) {
    int t = blockIdx.x * blockDim.x + threadIdx.x;
    if (t >= 256 * 256) return;
    const int o = t >> 8, k = t & 255;
    Wf[t] = (k < 128) ? (KE1i[o * 384 + k] + KE1i[o * 384 + 128 + k])
                      : KE1i[o * 384 + 128 + k];
}

// ---------------------------------------------------------------------------
extern "C" void kernel_launch(void* const* d_in, const int* in_sizes, int n_in,
                              void* d_out, int out_size, void* d_ws, size_t ws_size,
                              hipStream_t stream)
{
    const float* xn_in = (const float*)d_in[0];   // (1,64,N)
    const float* xe_in = (const float*)d_in[1];   // (1,64,E)
    const int*   iInd  = (const int*)d_in[2];
    const int*   jInd  = (const int*)d_in[3];
    const float* K1N   = (const float*)d_in[4];   // 128x64
    const float* K2N   = (const float*)d_in[5];   // 128x128
    const float* K1E   = (const float*)d_in[6];   // 128x64
    const float* K2E   = (const float*)d_in[7];   // 128x128
    const float* KNo   = (const float*)d_in[8];   // 64x128
    const float* KE1   = (const float*)d_in[9];   // 2x256x384
    const float* KE2   = (const float*)d_in[10];  // 2x128x256

    const int N = in_sizes[0] / 64;
    const int E = in_sizes[2];
    const int h = 128;

    float* ws = (float*)d_ws;
    size_t o = 0;
    float* xn_h  = ws + o; o += (size_t)h * N;   // node hidden (128xN)
    float* xe_h  = ws + o; o += (size_t)h * E;   // edge hidden (128xE)
    float* xnc   = ws + o; o += (size_t)h * N;   // edge_ave accumulator (128xN)
    float* W1f   = ws + o; o += 256 * 256;       // folded KE1
    float* stats = ws + o; o += 8;               // [sum,sumsq,mean,rstd]

    const int TB = 256;
    auto gg = [](int C) { return dim3((unsigned)((C + 63) / 64)); };

    // ---- xn = tensor_ln(K2N * relu(K1N * xn)) ----
    fillz<<<1, 64, 0, stream>>>(stats, 8);
    fused2_wmma<64, 128, 128, false, false, true><<<gg(N), TB, 0, stream>>>(
        K1N, K2N, xn_in, nullptr, nullptr, nullptr, N, 0, xn_h, nullptr, nullptr, stats);
    finalize_stats<<<1, 1, 0, stream>>>(stats, 1.f / ((float)h * (float)N));
    ln_apply<<<4096, TB, 0, stream>>>(xn_h, (size_t)h * N, stats);

    // ---- xe = tensor_ln(K2E * relu(K1E * xe)) ----
    fillz<<<1, 64, 0, stream>>>(stats, 8);
    fused2_wmma<64, 128, 128, false, false, true><<<gg(E), TB, 0, stream>>>(
        K1E, K2E, xe_in, nullptr, nullptr, nullptr, E, 0, xe_h, nullptr, nullptr, stats);
    finalize_stats<<<1, 1, 0, stream>>>(stats, 1.f / ((float)h * (float)E));
    ln_apply<<<8192, TB, 0, stream>>>(xe_h, (size_t)h * E, stats);

    // ---- L = 2 fused edge-conv layers ----
    for (int l = 0; l < 2; ++l) {
        fold_ke1<<<256, 256, 0, stream>>>(KE1 + (size_t)l * 256 * 384, W1f);
        fillz<<<4096, TB, 0, stream>>>(xnc, (size_t)h * N);
        // xec = KE2*relu(W1f*[xn[i];xe]); scatter 0.5*xec to xnc at i and j;
        // xe = relu(xe + xec)   -- all in one kernel, xec never hits HBM.
        fused2_wmma<256, 256, 128, true, true, false><<<gg(E), TB, 0, stream>>>(
            W1f, KE2 + (size_t)l * 128 * 256, xe_h, xn_h, iInd, jInd, E, N,
            nullptr, xe_h, xnc, nullptr);
        // xn = relu(xn + tensor_ln(xnc))
        fillz<<<1, 64, 0, stream>>>(stats, 8);
        reduce_stats<<<2048, TB, 0, stream>>>(xnc, (size_t)h * N, stats);
        finalize_stats<<<1, 1, 0, stream>>>(stats, 1.f / ((float)h * (float)N));
        node_update<<<4096, TB, 0, stream>>>(xn_h, xnc, (size_t)h * N, stats);
    }

    // ---- xn_out = KNout * xn_h -> d_out; then xe_h -> d_out tail ----
    float* out = (float*)d_out;
    gemm_wmma<128, 64><<<gg(N), TB, 0, stream>>>(KNo, xn_h, N, out);
    hipMemcpyAsync(out + (size_t)64 * N, xe_h, (size_t)h * E * sizeof(float),
                   hipMemcpyDeviceToDevice, stream);
}
